// HardLSTM_74199855005915
// MI455X (gfx1250) — matrix operand order repaired and checked by reading, not compiled, not run on hardware
//
#include <hip/hip_runtime.h>

#define T_SEQ 300
#define BATCH 32
#define DIN   320
#define HID   1024
#define NL    3
#define GH    (4 * HID)            // 4096
#define MROWS (T_SEQ * BATCH)      // 9600

typedef __bf16 bf16_t;
typedef bf16_t v16bf __attribute__((ext_vector_type(16)));
typedef bf16_t v8bf  __attribute__((ext_vector_type(8)));
typedef float  v8f   __attribute__((ext_vector_type(8)));

static __device__ inline v16bf frag_cat(v8bf lo, v8bf hi) {
    v16bf r;
#pragma unroll
    for (int i = 0; i < 8; ++i) { r[i] = lo[i]; r[i + 8] = hi[i]; }
    return r;
}

// A fragment: 16x32 bf16 tile of row-major [M,K] at (m0,k0).
// ISA layout: lanes 0-15 -> K chunks [k0, k0+8) & [k0+16, k0+24);
//             lanes 16-31 -> [k0+8, k0+16) & [k0+24, k0+32). Row = m0 + lane%16.
static __device__ inline v16bf load_a(const bf16_t* __restrict__ A, int K, int m0, int k0) {
    int lane = threadIdx.x & 31;
    int r = lane & 15, kh = lane >> 4;
    const bf16_t* p = A + (size_t)(m0 + r) * K + k0 + kh * 8;
    v8bf lo = *(const v8bf*)p;
    v8bf hi = *(const v8bf*)(p + 16);
    return frag_cat(lo, hi);
}

// B fragment: 32x16 tile, sourced from weight W stored [N,K] row-major (i.e. B^T).
// ISA layout: lane n = n0 + lane%16; lanes 0-15 hold K chunks [k0, k0+8) & [k0+16, k0+24); lanes 16-31 [k0+8, k0+16) & [k0+24, k0+32).
static __device__ inline v16bf load_b(const bf16_t* __restrict__ W, int K, int n0, int k0) {
    int lane = threadIdx.x & 31;
    int r = lane & 15, kh = lane >> 4;
    const bf16_t* p = W + (size_t)(n0 + r) * K + k0 + kh * 8;
    v8bf lo = *(const v8bf*)p;
    v8bf hi = *(const v8bf*)(p + 16);
    return frag_cat(lo, hi);
}

static __device__ inline v8f wmma_bf16(v16bf a, v16bf b, v8f c) {
    return __builtin_amdgcn_wmma_f32_16x16x32_bf16(false, a, false, b, (short)0, c, false, false);
}

// ---------------- elementwise f32 -> bf16 ----------------
__global__ void f32_to_bf16_kernel(const float* __restrict__ in, bf16_t* __restrict__ out, size_t n) {
    size_t i = (size_t)blockIdx.x * blockDim.x + threadIdx.x;
    if (i < n) out[i] = (bf16_t)in[i];
}

// ---------------- GX = A[M,K] * W[N,K]^T + (b_ih + b_hh) ----------------
// Block: 256 threads = 8 waves; block tile 64(M) x 256(N); wave tile 32(M) x 64(N).
__global__ __launch_bounds__(256) void gemm_bias_wmma(
    const bf16_t* __restrict__ A, const bf16_t* __restrict__ W,
    const float* __restrict__ bih, const float* __restrict__ bhh,
    float* __restrict__ C, int N, int K) {
    int wave = threadIdx.x >> 5;
    int lane = threadIdx.x & 31;
    int m0 = blockIdx.x * 64 + (wave & 1) * 32;
    int n0 = blockIdx.y * 256 + (wave >> 1) * 64;

    v8f acc[2][4];
#pragma unroll
    for (int mt = 0; mt < 2; ++mt)
#pragma unroll
        for (int nt = 0; nt < 4; ++nt) acc[mt][nt] = (v8f)0.0f;

    for (int k0 = 0; k0 < K; k0 += 32) {
        v16bf a0 = load_a(A, K, m0, k0);
        v16bf a1 = load_a(A, K, m0 + 16, k0);
#pragma unroll
        for (int nt = 0; nt < 4; ++nt) {
            v16bf b = load_b(W, K, n0 + nt * 16, k0);
            acc[0][nt] = wmma_bf16(a0, b, acc[0][nt]);
            acc[1][nt] = wmma_bf16(a1, b, acc[1][nt]);
        }
    }

    int r = lane & 15, half = lane >> 4;
#pragma unroll
    for (int mt = 0; mt < 2; ++mt) {
#pragma unroll
        for (int nt = 0; nt < 4; ++nt) {
            int n = n0 + nt * 16 + r;
            float bias = bih[n] + bhh[n];
            int mbase = m0 + mt * 16 + half * 8;
            float* cp = C + (size_t)mbase * N + n;
#pragma unroll
            for (int i = 0; i < 8; ++i) cp[(size_t)i * N] = acc[mt][nt][i] + bias;
        }
    }
}

// ---------------- per-layer state init (reference bug: c init = h0) ----------------
__global__ void lstm_init_kernel(const float* __restrict__ h0, bf16_t* __restrict__ hbf,
                                 float* __restrict__ hstate, float* __restrict__ cstate) {
    int i = blockIdx.x * 256 + threadIdx.x;   // 2*32*1024 total
    float v = h0[i];
    hbf[i] = (bf16_t)v;
    hstate[i] = v;
    cstate[i] = v;
}

// ---------------- one timestep, both directions fused ----------------
// grid.x = 32: blocks 0..15 -> dir 0 (forward, t=s), 16..31 -> dir 1 (backward, t=T-1-s).
// Block: 8 waves; wave owns (batch-half mt, 16 j-cols) and computes all 4 gate tiles.
__global__ __launch_bounds__(256) void lstm_step_kernel(
    const bf16_t* __restrict__ hbf_in,   // [2][B][H]
    bf16_t* __restrict__ hbf_out,        // [2][B][H]
    const bf16_t* __restrict__ whh,      // [2][4H][H]
    const float* __restrict__ gx0,       // [T][B][4H] dir 0
    const float* __restrict__ gx1,       // [T][B][4H] dir 1
    float* __restrict__ cstate,          // [2][B][H]
    float* __restrict__ hstate,          // [2][B][H]
    bf16_t* __restrict__ ybf,            // [T][B][2H]
    float* __restrict__ yout,            // [T][B][2H] or nullptr
    int s) {
    int d  = blockIdx.x >> 4;
    int jb = blockIdx.x & 15;
    int wave = threadIdx.x >> 5, lane = threadIdx.x & 31;
    int b0 = (wave & 1) * 16;
    int j0 = jb * 64 + (wave >> 1) * 16;
    int t = d ? (T_SEQ - 1 - s) : s;

    const bf16_t* hA = hbf_in + (size_t)d * BATCH * HID;
    const bf16_t* Wd = whh + (size_t)d * GH * HID;

    v8f acc[4];
#pragma unroll
    for (int g = 0; g < 4; ++g) acc[g] = (v8f)0.0f;

    for (int k0 = 0; k0 < HID; k0 += 32) {
        v16bf a = load_a(hA, HID, b0, k0);
#pragma unroll
        for (int g = 0; g < 4; ++g) {
            v16bf b = load_b(Wd, HID, g * HID + j0, k0);
            acc[g] = wmma_bf16(a, b, acc[g]);
        }
    }

    int j = j0 + (lane & 15);
    int bb0 = b0 + (lane >> 4) * 8;
    const float* gxd = (d ? gx1 : gx0) + (size_t)t * BATCH * GH;

#pragma unroll
    for (int i = 0; i < 8; ++i) {
        int bm = bb0 + i;
        const float* gr = gxd + (size_t)bm * GH;
        float gi = acc[0][i] + gr[j];
        float gf = acc[1][i] + gr[HID + j];
        float gg = acc[2][i] + gr[2 * HID + j];
        float go = acc[3][i] + gr[3 * HID + j];

        float ii = fminf(fmaxf(0.2f * gi + 0.5f, 0.0f), 1.0f);
        float ff = fminf(fmaxf(0.2f * gf + 0.5f, 0.0f), 1.0f);
        float gc = fminf(fmaxf(gg, -1.0f), 1.0f);
        float oo = fminf(fmaxf(0.2f * go + 0.5f, 0.0f), 1.0f);

        size_t ci = (size_t)(d * BATCH + bm) * HID + j;
        float c = ff * cstate[ci] + ii * gc;
        cstate[ci] = c;
        float h = oo * fminf(fmaxf(c, -1.0f), 1.0f);
        hstate[ci] = h;
        hbf_out[ci] = (bf16_t)h;

        size_t yi = ((size_t)t * BATCH + bm) * (2 * HID) + (size_t)d * HID + j;
        ybf[yi] = (bf16_t)h;
        if (yout) yout[yi] = h;
    }
}

__global__ void copy_hc_kernel(const float* __restrict__ hstate, const float* __restrict__ cstate,
                               float* __restrict__ out_h, float* __restrict__ out_c) {
    int i = blockIdx.x * 256 + threadIdx.x;   // 2*32*1024 total
    out_h[i] = hstate[i];
    out_c[i] = cstate[i];
}

extern "C" void kernel_launch(void* const* d_in, const int* in_sizes, int n_in,
                              void* d_out, int out_size, void* d_ws, size_t ws_size,
                              hipStream_t stream) {
    (void)in_sizes; (void)n_in; (void)out_size; (void)ws_size;

    const float* x     = (const float*)d_in[0];
    const float* h0    = (const float*)d_in[1];
    // d_in[2] (c0) intentionally unused: reference initializes cell state from h0.
    const float* w_ih0 = (const float*)d_in[3];
    const float* w_hh0 = (const float*)d_in[4];
    const float* b_ih0 = (const float*)d_in[5];
    const float* b_hh0 = (const float*)d_in[6];
    const float* w_ihL = (const float*)d_in[7];
    const float* w_hhL = (const float*)d_in[8];
    const float* b_ihL = (const float*)d_in[9];
    const float* b_hhL = (const float*)d_in[10];
    float* out = (float*)d_out;

    char* ws = (char*)d_ws;
    size_t off = 0;
    auto alloc = [&](size_t bytes) -> void* {
        void* p = (void*)(ws + off);
        off += (bytes + 255) & ~(size_t)255;
        return p;
    };

    bf16_t* wih0b = (bf16_t*)alloc((size_t)2 * GH * DIN * 2);
    bf16_t* wihLb = (bf16_t*)alloc((size_t)4 * GH * 2 * HID * 2);
    bf16_t* whhb  = (bf16_t*)alloc((size_t)6 * GH * HID * 2);
    bf16_t* xbf   = (bf16_t*)alloc((size_t)MROWS * DIN * 2);
    bf16_t* ybf0  = (bf16_t*)alloc((size_t)MROWS * 2 * HID * 2);
    bf16_t* ybf1  = (bf16_t*)alloc((size_t)MROWS * 2 * HID * 2);
    float*  gx0   = (float*)alloc((size_t)MROWS * GH * 4);
    float*  gx1   = (float*)alloc((size_t)MROWS * GH * 4);
    bf16_t* hbfA  = (bf16_t*)alloc((size_t)2 * BATCH * HID * 2);
    bf16_t* hbfB  = (bf16_t*)alloc((size_t)2 * BATCH * HID * 2);
    float*  cst   = (float*)alloc((size_t)2 * BATCH * HID * 4);
    float*  hst   = (float*)alloc((size_t)2 * BATCH * HID * 4);

    auto conv = [&](const float* src, bf16_t* dst, size_t n) {
        int blocks = (int)((n + 255) / 256);
        f32_to_bf16_kernel<<<blocks, 256, 0, stream>>>(src, dst, n);
    };
    conv(w_ih0, wih0b, (size_t)2 * GH * DIN);
    conv(w_ihL, wihLb, (size_t)4 * GH * 2 * HID);
    conv(w_hh0, whhb, (size_t)2 * GH * HID);                      // layer 0 -> slots 0,1
    conv(w_hhL, whhb + (size_t)2 * GH * HID, (size_t)4 * GH * HID); // layers 1,2 -> slots 2..5
    conv(x, xbf, (size_t)MROWS * DIN);

    const size_t OUT_Y = (size_t)MROWS * 2 * HID;   // 19,660,800
    const size_t HN_SZ = (size_t)6 * BATCH * HID;

    for (int l = 0; l < NL; ++l) {
        const bf16_t* inbf = (l == 0) ? xbf : (l == 1 ? ybf0 : ybf1);
        int K = (l == 0) ? DIN : 2 * HID;
        bf16_t* youtbf = (l == 0) ? ybf0 : (l == 1 ? ybf1 : ybf0 /* scratch */);
        float* yf32 = (l == NL - 1) ? out : nullptr;

        for (int d = 0; d < 2; ++d) {
            const bf16_t* wb = (l == 0)
                ? wih0b + (size_t)d * GH * DIN
                : wihLb + (size_t)((l - 1) * 2 + d) * GH * 2 * HID;
            const float* bi = (l == 0) ? b_ih0 + (size_t)d * GH
                                       : b_ihL + (size_t)((l - 1) * 2 + d) * GH;
            const float* bh = (l == 0) ? b_hh0 + (size_t)d * GH
                                       : b_hhL + (size_t)((l - 1) * 2 + d) * GH;
            float* gx = (d == 0) ? gx0 : gx1;
            dim3 grid(MROWS / 64, GH / 256);
            gemm_bias_wmma<<<grid, 256, 0, stream>>>(inbf, wb, bi, bh, gx, GH, K);
        }

        lstm_init_kernel<<<(2 * BATCH * HID) / 256, 256, 0, stream>>>(
            h0 + (size_t)l * 2 * BATCH * HID, hbfA, hst, cst);

        const bf16_t* whh_l = whhb + (size_t)l * 2 * GH * HID;
        for (int s = 0; s < T_SEQ; ++s) {
            const bf16_t* hin = (s & 1) ? hbfB : hbfA;
            bf16_t* hout = (s & 1) ? hbfA : hbfB;
            lstm_step_kernel<<<32, 256, 0, stream>>>(hin, hout, whh_l, gx0, gx1,
                                                     cst, hst, youtbf, yf32, s);
        }

        copy_hc_kernel<<<(2 * BATCH * HID) / 256, 256, 0, stream>>>(
            hst, cst,
            out + OUT_Y + (size_t)l * 2 * BATCH * HID,
            out + OUT_Y + HN_SZ + (size_t)l * 2 * BATCH * HID);
    }
}
